// EquivariantBlock_72060961292773
// MI455X (gfx1250) — matrix operand order repaired and checked by reading, not compiled, not run on hardware
//
#include <hip/hip_runtime.h>
#include <hip/hip_bf16.h>

// ---------------------------------------------------------------------------
// EGNN EquivariantBlock for MI455X (gfx1250, wave32, WMMA bf16 16x16x32).
// N=20000 nodes, E=640000 edges, H=128.
// Edge MLPs fused end-to-end; 64-edge tiles x 8 waves: each B fragment loaded
// from L2 feeds 4 WMMAs (weight reuse across M-tiles).
// ---------------------------------------------------------------------------

#define NN 20000
#define EE 640000
#define HH 128

typedef float  v8f   __attribute__((ext_vector_type(8)));
typedef __bf16 v16bf __attribute__((ext_vector_type(16)));

__device__ __forceinline__ float silu_f(float x) { return x / (1.0f + __expf(-x)); }
__device__ __forceinline__ float sigm_f(float x) { return 1.0f / (1.0f + __expf(-x)); }

__device__ __forceinline__ v8f wmma_bf16(v16bf a, v16bf b, v8f c) {
    return __builtin_amdgcn_wmma_f32_16x16x32_bf16(
        false, a, false, b, (short)0, c, false, false);
}

// A-fragment (16x32 bf16) from LDS, row-major [16][krow] starting at `lds`.
__device__ __forceinline__ v16bf load_a_frag(const __bf16* lds, int krow, int kt) {
    int lane = threadIdx.x & 31;
    int m  = lane & 15;
    int hi = lane >> 4;
    const __bf16* p = lds + m * krow + kt * 32 + hi * 8;
    union { uint4 u[2]; v16bf v; } f;
    f.u[0] = *(const uint4*)(p);        // ds_load_b128
    f.u[1] = *(const uint4*)(p + 16);
    return f.v;
}

// B-fragment (32x16 bf16) pre-packed in fragment order: 32B contiguous per lane.
__device__ __forceinline__ v16bf load_b_frag(const __bf16* Wp, int kt, int nt) {
    int lane = threadIdx.x & 31;
    return *(const v16bf*)(Wp + (((size_t)(kt * 8 + nt) * 32 + lane) << 4));
}

// Pack fp32 [K,128] row-major weight into bf16 WMMA B-fragment order.
__global__ void pack_weight_bf16(const float* __restrict__ W, __bf16* __restrict__ Wp, int K) {
    int tid = blockIdx.x * blockDim.x + threadIdx.x;
    if (tid >= K * 128) return;
    int j    = tid & 15;
    int lane = (tid >> 4) & 31;
    int frag = tid >> 9;
    int kt = frag >> 3, nt = frag & 7;
    int hi = lane >> 4;
    int k = kt * 32 + hi * 8 + (j & 7) + ((j >> 3) << 4);
    int n = nt * 16 + (lane & 15);
    Wp[tid] = (__bf16)W[k * 128 + n];
}

__global__ void zero_f32(float* __restrict__ p, int count) {
    int t = blockIdx.x * blockDim.x + threadIdx.x;
    if (t < count) p[t] = 0.0f;
}

// coord2diff: radial + normalized coord_diff per edge
__global__ void coord_prep(const float* __restrict__ x, const int* __restrict__ eidx,
                           float* __restrict__ radial, float* __restrict__ cdiff) {
    int e = blockIdx.x * blockDim.x + threadIdx.x;
    if (e >= EE) return;
    int r = eidx[e], c = eidx[EE + e];
    float dx = x[r * 3 + 0] - x[c * 3 + 0];
    float dy = x[r * 3 + 1] - x[c * 3 + 1];
    float dz = x[r * 3 + 2] - x[c * 3 + 2];
    float rad = dx * dx + dy * dy + dz * dz;
    float inv = 1.0f / (sqrtf(rad + 1e-8f) + 1.0f);   // NORM_CONST = 1
    radial[e] = rad;
    cdiff[e * 3 + 0] = dx * inv;
    cdiff[e * 3 + 1] = dy * inv;
    cdiff[e * 3 + 2] = dz * inv;
}

// ---------------------------------------------------------------------------
// Fused edge MLP, 64 edges / block, 256 threads (8 waves).
// Wave w owns N-tile w and iterates 4 M-tiles -> 4x B-fragment reuse.
// EQUIV=false: GCL edge model -> att -> scatter into agg[N,128].
// EQUIV=true : coord model -> scalar cm -> scatter into xagg[N,3].
// ---------------------------------------------------------------------------
template <bool EQUIV>
__global__ __launch_bounds__(256) void edge_kernel(
    const float*  __restrict__ hbuf,
    const int*    __restrict__ eidx,
    const float*  __restrict__ radial,
    const float*  __restrict__ eattr,
    const float*  __restrict__ emask,
    const __bf16* __restrict__ W1p,     // packed [256,128]
    const __bf16* __restrict__ W2p,     // packed [128,128]
    const float*  __restrict__ W1raw,   // fp32 [258,128] (rows 256,257 used)
    const float*  __restrict__ b1,
    const float*  __restrict__ b2,
    const float*  __restrict__ headW,   // aW [128] (GCL) or W3 [128] (EQUIV)
    const float*  __restrict__ headB,   // ab [1]  (GCL) or unused
    const float*  __restrict__ cdiff,   // [E,3] (EQUIV only)
    float*        __restrict__ out)     // agg [N,128] or xagg [N,3]
{
    __shared__ __align__(16) __bf16 sIn[64 * 256];   // 32KB; aliased as mij f32 later
    __shared__ __align__(16) __bf16 sHid[64 * 128];  // 16KB
    __shared__ float  sEa0[64], sEa1[64], sEm[64], sAtt[64];
    __shared__ int    sRow[64];
    float* sMijF = reinterpret_cast<float*>(sIn);    // [64][128] f32 (32KB)

    const int t = threadIdx.x;
    const int tile = blockIdx.x * 64;                 // E is a multiple of 64

    if (t < 64) {
        int e = tile + t;
        sRow[t] = eidx[e];
        sEa0[t] = radial[e];
        sEa1[t] = eattr[e];
        sEm[t]  = emask[e];
    }
    {   // gather h[row] || h[col] -> LDS bf16 [64][256]; 64 floats per thread
        int e = t >> 2, seg = t & 3;                  // seg: 0,1=h_row halves; 2,3=h_col halves
        int eg = tile + e;
        int nsrc = (seg < 2) ? eidx[eg] : eidx[EE + eg];
        int cbase = (seg & 1) * 64;
        const float* src = hbuf + (size_t)nsrc * HH + cbase;
        __bf16* dst = sIn + e * 256 + ((seg < 2) ? 0 : 128) + cbase;
        #pragma unroll
        for (int i = 0; i < 64; i += 4) {
            float4 v = *(const float4*)(src + i);
            dst[i + 0] = (__bf16)v.x; dst[i + 1] = (__bf16)v.y;
            dst[i + 2] = (__bf16)v.z; dst[i + 3] = (__bf16)v.w;
        }
    }
    __syncthreads();

    const int lane = t & 31;
    const int nt = t >> 5;                            // wave id == N-tile
    const int laneN = lane & 15, hi = lane >> 4;
    const int n = nt * 16 + laneN;

    // ---- GEMM1: [64,256] x [256,128] (K=256, 8 k-steps, 4 M-tiles) ----
    v8f acc[4] = { v8f{}, v8f{}, v8f{}, v8f{} };
    #pragma unroll
    for (int kt = 0; kt < 8; ++kt) {
        v16bf b = load_b_frag(W1p, kt, nt);
        #pragma unroll
        for (int mt = 0; mt < 4; ++mt)
            acc[mt] = wmma_bf16(load_a_frag(sIn + mt * 16 * 256, 256, kt), b, acc[mt]);
    }
    // epilogue: + bias + rank-2 update (radial, edge_attr rows) -> SiLU -> LDS bf16
    {
        float bb = b1[n];
        float wA = W1raw[256 * 128 + n];
        float wB = W1raw[257 * 128 + n];
        #pragma unroll
        for (int mt = 0; mt < 4; ++mt)
            #pragma unroll
            for (int r = 0; r < 8; ++r) {
                int m = mt * 16 + r + hi * 8;
                float v = acc[mt][r] + bb + sEa0[m] * wA + sEa1[m] * wB;
                sHid[m * 128 + n] = (__bf16)silu_f(v);
            }
    }
    __syncthreads();

    // ---- GEMM2: [64,128] x [128,128] (K=128, 4 k-steps, 4 M-tiles) ----
    acc[0] = v8f{}; acc[1] = v8f{}; acc[2] = v8f{}; acc[3] = v8f{};
    #pragma unroll
    for (int kt = 0; kt < 4; ++kt) {
        v16bf b = load_b_frag(W2p, kt, nt);
        #pragma unroll
        for (int mt = 0; mt < 4; ++mt)
            acc[mt] = wmma_bf16(load_a_frag(sHid + mt * 16 * 128, 128, kt), b, acc[mt]);
    }
    {
        float bb = b2[n];
        #pragma unroll
        for (int mt = 0; mt < 4; ++mt)
            #pragma unroll
            for (int r = 0; r < 8; ++r) {
                int m = mt * 16 + r + hi * 8;
                sMijF[m * 128 + n] = silu_f(acc[mt][r] + bb); // mij / pre-W3 hidden
            }
    }
    __syncthreads();

    if (!EQUIV) {
        if (t < 64) {   // att = sigmoid(mij . aW + ab)
            float s = headB[0];
            const float* mr = sMijF + t * 128;
            #pragma unroll
            for (int c = 0; c < 128; c += 4) {
                float4 mv = *(const float4*)(mr + c);
                float4 wv = *(const float4*)(headW + c);
                s += mv.x * wv.x + mv.y * wv.y + mv.z * wv.z + mv.w * wv.w;
            }
            sAtt[t] = sigm_f(s);
        }
        __syncthreads();
        {   // scatter: agg[row] += mij * att * edge_mask  (32 atomics / thread)
            int e = t >> 2, cb = (t & 3) * 32;
            float a = sAtt[e] * sEm[e];
            float* dst = out + (size_t)sRow[e] * HH + cb;
            const float* mr = sMijF + e * 128 + cb;
            #pragma unroll
            for (int i = 0; i < 32; ++i)
                atomicAdd(&dst[i], mr[i] * a);
        }
    } else {
        if (t < 64) {   // cm = hidden . W3 ; scatter coord update
            float s = 0.0f;
            const float* mr = sMijF + t * 128;
            #pragma unroll
            for (int c = 0; c < 128; c += 4) {
                float4 mv = *(const float4*)(mr + c);
                float4 wv = *(const float4*)(headW + c);
                s += mv.x * wv.x + mv.y * wv.y + mv.z * wv.z + mv.w * wv.w;
            }
            float mlt = s * sEm[t];
            int eg = tile + t;
            float* dst = out + (size_t)sRow[t] * 3;
            atomicAdd(&dst[0], cdiff[eg * 3 + 0] * mlt);
            atomicAdd(&dst[1], cdiff[eg * 3 + 1] * mlt);
            atomicAdd(&dst[2], cdiff[eg * 3 + 2] * mlt);
        }
    }
}

// ---------------------------------------------------------------------------
// Node model: h = node_mask * (h + MLP([h, agg/100]))  (16 nodes / block)
// ---------------------------------------------------------------------------
__global__ __launch_bounds__(128) void node_kernel(
    float*        __restrict__ hbuf,
    const float*  __restrict__ agg,
    const __bf16* __restrict__ W1p,   // packed [256,128]
    const __bf16* __restrict__ W2p,   // packed [128,128]
    const float*  __restrict__ b1,
    const float*  __restrict__ b2,
    const float*  __restrict__ nmask)
{
    __shared__ __align__(16) __bf16 sIn[16 * 256];
    __shared__ __align__(16) __bf16 sHid[16 * 128];

    const int t = threadIdx.x;
    const int tile = blockIdx.x * 16;                 // N multiple of 16

    {   // gather [h, agg*0.01] -> LDS bf16
        int nd = t >> 3, seg = t & 7;
        int g = tile + nd;
        int cbase = (seg & 3) * 32;
        const float* src = (seg < 4) ? (hbuf + (size_t)g * HH + cbase)
                                     : (agg  + (size_t)g * HH + cbase);
        float sc = (seg < 4) ? 1.0f : 0.01f;          // / NORM_FACTOR
        __bf16* dst = sIn + nd * 256 + ((seg < 4) ? 0 : 128) + cbase;
        #pragma unroll
        for (int i = 0; i < 32; i += 4) {
            float4 v = *(const float4*)(src + i);
            dst[i + 0] = (__bf16)(v.x * sc); dst[i + 1] = (__bf16)(v.y * sc);
            dst[i + 2] = (__bf16)(v.z * sc); dst[i + 3] = (__bf16)(v.w * sc);
        }
    }
    __syncthreads();

    const int wave = t >> 5, lane = t & 31;
    const int laneN = lane & 15, hi = lane >> 4;
    const int ntv[2] = { wave * 2, wave * 2 + 1 };

    v8f acc[2] = { v8f{}, v8f{} };
    #pragma unroll
    for (int kt = 0; kt < 8; ++kt) {
        v16bf a = load_a_frag(sIn, 256, kt);
        #pragma unroll
        for (int p = 0; p < 2; ++p)
            acc[p] = wmma_bf16(a, load_b_frag(W1p, kt, ntv[p]), acc[p]);
    }
    #pragma unroll
    for (int p = 0; p < 2; ++p) {
        int n = ntv[p] * 16 + laneN;
        float bb = b1[n];
        #pragma unroll
        for (int r = 0; r < 8; ++r) {
            int m = r + hi * 8;
            sHid[m * 128 + n] = (__bf16)silu_f(acc[p][r] + bb);
        }
    }
    __syncthreads();

    acc[0] = v8f{}; acc[1] = v8f{};
    #pragma unroll
    for (int kt = 0; kt < 4; ++kt) {
        v16bf a = load_a_frag(sHid, 128, kt);
        #pragma unroll
        for (int p = 0; p < 2; ++p)
            acc[p] = wmma_bf16(a, load_b_frag(W2p, kt, ntv[p]), acc[p]);
    }
    // residual + node_mask, in-place update (each thread rewrites cells it read)
    #pragma unroll
    for (int p = 0; p < 2; ++p) {
        int n = ntv[p] * 16 + laneN;
        float bb = b2[n];
        #pragma unroll
        for (int r = 0; r < 8; ++r) {
            int m = r + hi * 8;
            int g = tile + m;
            float hv = hbuf[(size_t)g * HH + n];
            hbuf[(size_t)g * HH + n] = (hv + acc[p][r] + bb) * nmask[g];
        }
    }
}

// h_out = h*node_mask ; x_out = (x + xagg/100)*node_mask
__global__ void finalize_kernel(const float* __restrict__ hbuf, const float* __restrict__ x,
                                const float* __restrict__ xagg, const float* __restrict__ nmask,
                                float* __restrict__ out) {
    int t = blockIdx.x * blockDim.x + threadIdx.x;
    const int HTOT = NN * HH;
    if (t < HTOT) {
        out[t] = hbuf[t] * nmask[t >> 7];
    } else if (t < HTOT + NN * 3) {
        int i = t - HTOT;
        int nd = i / 3;
        out[t] = (x[i] + xagg[i] * 0.01f) * nmask[nd];
    }
}

// ---------------------------------------------------------------------------
extern "C" void kernel_launch(void* const* d_in, const int* in_sizes, int n_in,
                              void* d_out, int out_size, void* d_ws, size_t ws_size,
                              hipStream_t stream) {
    // pytree-flatten order: h,x,edge_index,edge_attr,node_mask,edge_mask,
    // equiv{W1,W2,W3,b1,b2}, gcl0{aW,ab,eW1,eW2,eb1,eb2,nW1,nW2,nb1,nb2}, gcl1{...}
    const float* h_in  = (const float*)d_in[0];
    const float* x_in  = (const float*)d_in[1];
    const int*   eidx  = (const int*)d_in[2];
    const float* eattr = (const float*)d_in[3];
    const float* nmask = (const float*)d_in[4];
    const float* emask = (const float*)d_in[5];
    const float* qW1 = (const float*)d_in[6];
    const float* qW2 = (const float*)d_in[7];
    const float* qW3 = (const float*)d_in[8];
    const float* qb1 = (const float*)d_in[9];
    const float* qb2 = (const float*)d_in[10];
    const float* g[2][10];
    for (int L = 0; L < 2; ++L)
        for (int i = 0; i < 10; ++i)
            g[L][i] = (const float*)d_in[11 + L * 10 + i];
    // g[L]: 0=aW 1=ab 2=eW1 3=eW2 4=eb1 5=eb2 6=nW1 7=nW2 8=nb1 9=nb2

    char* ws = (char*)d_ws;
    size_t cur = 0;
    auto take = [&](size_t bytes) -> char* {
        char* p = ws + cur;
        cur += (bytes + 255) & ~(size_t)255;
        return p;
    };
    float*  hbuf   = (float*)take((size_t)NN * HH * 4);
    float*  agg    = (float*)take((size_t)NN * HH * 4);
    float*  xagg   = (float*)take((size_t)NN * 3 * 4);
    float*  radial = (float*)take((size_t)EE * 4);
    float*  cdiff  = (float*)take((size_t)EE * 3 * 4);
    __bf16* eW1p[2], *eW2p[2], *nW1p[2], *nW2p[2];
    for (int L = 0; L < 2; ++L) {
        eW1p[L] = (__bf16*)take(256 * 128 * 2);
        eW2p[L] = (__bf16*)take(128 * 128 * 2);
        nW1p[L] = (__bf16*)take(256 * 128 * 2);
        nW2p[L] = (__bf16*)take(128 * 128 * 2);
    }
    __bf16* qW1p = (__bf16*)take(256 * 128 * 2);
    __bf16* qW2p = (__bf16*)take(128 * 128 * 2);
    (void)ws_size; (void)in_sizes; (void)n_in; (void)out_size;

    // ---- prep ----
    hipMemcpyAsync(hbuf, h_in, (size_t)NN * HH * 4, hipMemcpyDeviceToDevice, stream);
    coord_prep<<<(EE + 255) / 256, 256, 0, stream>>>(x_in, eidx, radial, cdiff);
    const int PK256 = (256 * 128 + 255) / 256, PK128 = (128 * 128 + 255) / 256;
    for (int L = 0; L < 2; ++L) {
        pack_weight_bf16<<<PK256, 256, 0, stream>>>(g[L][2], eW1p[L], 256); // eW1[0:256]
        pack_weight_bf16<<<PK128, 256, 0, stream>>>(g[L][3], eW2p[L], 128);
        pack_weight_bf16<<<PK256, 256, 0, stream>>>(g[L][6], nW1p[L], 256);
        pack_weight_bf16<<<PK128, 256, 0, stream>>>(g[L][7], nW2p[L], 128);
    }
    pack_weight_bf16<<<PK256, 256, 0, stream>>>(qW1, qW1p, 256);
    pack_weight_bf16<<<PK128, 256, 0, stream>>>(qW2, qW2p, 128);

    // ---- 2 GCL layers ----
    for (int L = 0; L < 2; ++L) {
        zero_f32<<<(NN * HH + 255) / 256, 256, 0, stream>>>(agg, NN * HH);
        edge_kernel<false><<<EE / 64, 256, 0, stream>>>(
            hbuf, eidx, radial, eattr, emask,
            eW1p[L], eW2p[L], g[L][2], g[L][4], g[L][5],
            g[L][0], g[L][1], nullptr, agg);
        node_kernel<<<NN / 16, 128, 0, stream>>>(
            hbuf, agg, nW1p[L], nW2p[L], g[L][8], g[L][9], nmask);
    }

    // ---- equivariant coord update ----
    zero_f32<<<(NN * 3 + 255) / 256, 256, 0, stream>>>(xagg, NN * 3);
    edge_kernel<true><<<EE / 64, 256, 0, stream>>>(
        hbuf, eidx, radial, eattr, emask,
        qW1p, qW2p, qW1, qb1, qb2,
        qW3, nullptr, cdiff, xagg);

    finalize_kernel<<<(NN * HH + NN * 3 + 255) / 256, 256, 0, stream>>>(
        hbuf, x_in, xagg, nmask, (float*)d_out);
}
